// DiSAN_25185688223842
// MI455X (gfx1250) — matrix operand order repaired and checked by reading, not compiled, not run on hardware
//
#include <hip/hip_runtime.h>
#include <hip/hip_bf16.h>
#include <stdint.h>

// DiSAN forward for MI455X (gfx1250, wave32, WMMA).
// B=16, L=128, D=256, V=32000. All GEMMs run on v_wmma_f32_16x16x32_bf16
// with pre-converted bf16 operands and fully vectorized b128 fragment loads.

#define Bdim 16
#define Ldim 128
#define Ddim 256
#define Mrows (Bdim * Ldim)   // 2048

typedef __attribute__((ext_vector_type(16))) __bf16 v16bf;
typedef __attribute__((ext_vector_type(8)))  __bf16 v8bf;
typedef __attribute__((ext_vector_type(8)))  float  v8f;

#define EPI_NONE     0
#define EPI_BIAS     1
#define EPI_BIAS_ELU 2
#define EPI_FUSE_F   3

// ---------------------------------------------------------------------------
// Setup: W (KxN fp32, row-major) -> WT (NxK bf16).  One thread per element;
// reads coalesced over n. Weights are tiny (<=1MB) and L2-resident after this.
// ---------------------------------------------------------------------------
__global__ __launch_bounds__(256) void transpose_bf16(
    const float* __restrict__ W, __bf16* __restrict__ WT, int K, int N)
{
  const int idx = blockIdx.x * 256 + threadIdx.x;
  const int k = idx / N, n = idx % N;
  WT[(size_t)n * K + k] = (__bf16)W[idx];
}

// ebf[r, :] = bf16(emb[x[r], :])
__global__ __launch_bounds__(256) void gather_bf16(
    const int* __restrict__ x, const float* __restrict__ emb,
    __bf16* __restrict__ ebf)
{
  const int r = blockIdx.x, d = threadIdx.x;
  ebf[(size_t)r * Ddim + d] = (__bf16)emb[(size_t)x[r] * Ddim + d];
}

// ---------------------------------------------------------------------------
// WMMA GEMM: out = epilogue(A(MxK,bf16) @ WT^T),  WT is NxK bf16.
// Each wave owns a 16x64 C tile (4 fragments, A fragment reused 4x);
// 8 waves per 256-thread block. Fragment layouts (A/B: this code's contiguous k order, not ISA table order):
//   A 16x32: lane holds 16 consecutive k for row m = l16; khalf = lane>>4
//   B 32x16: lane holds 16 consecutive k for column n = l16; khalf = lane>>4
//   C 16x16: VGPR r -> row = half*8 + r, col = l16
// All fragment loads are 16-byte (b128).
// ---------------------------------------------------------------------------
template <int EPI, int K>
__global__ __launch_bounds__(256) void gemm_wmma(
    const __bf16* __restrict__ A, const __bf16* __restrict__ WT,
    const float* __restrict__ bias, const float* __restrict__ addend,
    const float* __restrict__ hbuf, const float* __restrict__ sbuf,
    float* __restrict__ out, __bf16* __restrict__ outbf,
    int M, int N, int ldout, int colOff)
{
  const int lane   = threadIdx.x & 31;
  const int wave   = threadIdx.x >> 5;
  const int tilesN = N >> 6;
  const int gw = blockIdx.x * 8 + wave;
  const int mt = (gw / tilesN) << 4;
  const int nt = (gw % tilesN) << 6;
  if (mt >= M) return;  // uniform per wave; EXEC stays all-1s for WMMA

  const int half = lane >> 4;
  const int l16  = lane & 15;

  const __bf16* arow = A + (size_t)(mt + l16) * K;
  const __bf16* wrow0 = WT + (size_t)(nt + l16) * K + half * 16;
  const __bf16* wrow1 = wrow0 + (size_t)16 * K;
  const __bf16* wrow2 = wrow0 + (size_t)32 * K;
  const __bf16* wrow3 = wrow0 + (size_t)48 * K;

  v8f acc[4] = {v8f{}, v8f{}, v8f{}, v8f{}};

#pragma unroll
  for (int k0 = 0; k0 < K; k0 += 32) {
    const v8bf alo = *(const v8bf*)(arow + k0 + half * 16);
    const v8bf ahi = *(const v8bf*)(arow + k0 + half * 16 + 8);
    const v16bf af = __builtin_shufflevector(
        alo, ahi, 0, 1, 2, 3, 4, 5, 6, 7, 8, 9, 10, 11, 12, 13, 14, 15);

    const v8bf b0l = *(const v8bf*)(wrow0 + k0);
    const v8bf b0h = *(const v8bf*)(wrow0 + k0 + 8);
    const v8bf b1l = *(const v8bf*)(wrow1 + k0);
    const v8bf b1h = *(const v8bf*)(wrow1 + k0 + 8);
    const v8bf b2l = *(const v8bf*)(wrow2 + k0);
    const v8bf b2h = *(const v8bf*)(wrow2 + k0 + 8);
    const v8bf b3l = *(const v8bf*)(wrow3 + k0);
    const v8bf b3h = *(const v8bf*)(wrow3 + k0 + 8);

    const v16bf bf0 = __builtin_shufflevector(
        b0l, b0h, 0, 1, 2, 3, 4, 5, 6, 7, 8, 9, 10, 11, 12, 13, 14, 15);
    const v16bf bf1 = __builtin_shufflevector(
        b1l, b1h, 0, 1, 2, 3, 4, 5, 6, 7, 8, 9, 10, 11, 12, 13, 14, 15);
    const v16bf bf2 = __builtin_shufflevector(
        b2l, b2h, 0, 1, 2, 3, 4, 5, 6, 7, 8, 9, 10, 11, 12, 13, 14, 15);
    const v16bf bf3 = __builtin_shufflevector(
        b3l, b3h, 0, 1, 2, 3, 4, 5, 6, 7, 8, 9, 10, 11, 12, 13, 14, 15);

    acc[0] = __builtin_amdgcn_wmma_f32_16x16x32_bf16(
        false, af, false, bf0, (short)0, acc[0], false, false);
    acc[1] = __builtin_amdgcn_wmma_f32_16x16x32_bf16(
        false, af, false, bf1, (short)0, acc[1], false, false);
    acc[2] = __builtin_amdgcn_wmma_f32_16x16x32_bf16(
        false, af, false, bf2, (short)0, acc[2], false, false);
    acc[3] = __builtin_amdgcn_wmma_f32_16x16x32_bf16(
        false, af, false, bf3, (short)0, acc[3], false, false);
  }

#pragma unroll
  for (int t = 0; t < 4; ++t) {
#pragma unroll
    for (int r = 0; r < 8; ++r) {
      const int row = mt + half * 8 + r;
      const int col = nt + t * 16 + l16;
      float v = acc[t][r];
      if (EPI == EPI_BIAS || EPI == EPI_BIAS_ELU) v += bias[col];
      if (EPI == EPI_BIAS_ELU) v = (v > 0.f) ? v : expm1f(v);
      if (EPI == EPI_FUSE_F) {
        // f = sigmoid(s@Wf1 + h@Wf2 + b);  u = f*h + (1-f)*s
        v += addend[(size_t)row * N + col] + bias[col];
        const float f  = 1.f / (1.f + __expf(-v));
        const float hh = hbuf[(size_t)row * N + col];
        const float ss = sbuf[(size_t)row * N + col];
        v = f * hh + (1.f - f) * ss;
      }
      const size_t oidx = (size_t)row * ldout + colOff + col;
      out[oidx] = v;
      if (outbf) outbf[oidx] = (__bf16)v;
    }
  }
}

// ---------------------------------------------------------------------------
// Fused directional attention (per-feature softmax over m, online/flash).
// One block per (b,l); thread e in [0,256). tanh computed once per (m,e),
// two masked accumulators (fw masks m<=l, bw masks m>=l). -INF is a finite
// -1e13 so all-masked rows degrade to a uniform softmax exactly like jax.
// ---------------------------------------------------------------------------
__global__ __launch_bounds__(256) void disan_attn(
    const float* __restrict__ h, const float* __restrict__ h1,
    const float* __restrict__ h2, const float* __restrict__ bvec,
    const float* __restrict__ cptr, const unsigned char* __restrict__ mask,
    float* __restrict__ sfw, float* __restrict__ sbw,
    __bf16* __restrict__ sfwbf, __bf16* __restrict__ sbwbf)
{
  const int bl = blockIdx.x;
  const int b  = bl >> 7;      // / L
  const int l  = bl & 127;     // % L
  const int e  = threadIdx.x;

  __shared__ unsigned char smask[Ldim];
  if (threadIdx.x < Ldim) smask[threadIdx.x] = mask[b * Ldim + threadIdx.x];
  __syncthreads();

  const float cc   = cptr[0];
  const float invc = 1.f / cc;
  const float base = h1[(size_t)bl * Ddim + e] + bvec[e];
  const bool  padl = smask[l] != 0;
  const float NEG  = -1e13f;

  float Mf = -1e30f, Sf = 0.f, Wf = 0.f;
  float Mb = -1e30f, Sb = 0.f, Wb = 0.f;

  for (int m = 0; m < Ldim; ++m) {
    const size_t idx = (size_t)(b * Ldim + m) * Ddim + e;
    const float v  = cc * tanhf((base + h2[idx]) * invc);
    const float hv = h[idx];
    const bool pad = padl || (smask[m] != 0);
    const float vf = (pad || (m <= l)) ? NEG : v;
    const float vb = (pad || (m >= l)) ? NEG : v;
    // forward
    float nM = fmaxf(Mf, vf);
    float sc = __expf(Mf - nM), ev = __expf(vf - nM);
    Sf = Sf * sc + ev; Wf = Wf * sc + ev * hv; Mf = nM;
    // backward
    nM = fmaxf(Mb, vb);
    sc = __expf(Mb - nM); ev = __expf(vb - nM);
    Sb = Sb * sc + ev; Wb = Wb * sc + ev * hv; Mb = nM;
  }
  const size_t o = (size_t)bl * Ddim + e;
  const float rf = Wf / Sf, rb = Wb / Sb;
  sfw[o] = rf; sbw[o] = rb;
  sfwbf[o] = (__bf16)rf; sbwbf[o] = (__bf16)rb;
}

// out[b,n] = sum_l u[b,l,n] * att_s[b,l,n]   (n in [0,512))
__global__ __launch_bounds__(512) void disan_reduce(
    const float* __restrict__ u, const float* __restrict__ as,
    float* __restrict__ out)
{
  const int b = blockIdx.x;
  const int n = threadIdx.x;
  float acc = 0.f;
  for (int l = 0; l < Ldim; ++l) {
    const size_t idx = (size_t)(b * Ldim + l) * 512 + n;
    acc += u[idx] * as[idx];
  }
  out[b * 512 + n] = acc;
}

// ---------------------------------------------------------------------------
extern "C" void kernel_launch(void* const* d_in, const int* in_sizes, int n_in,
                              void* d_out, int out_size, void* d_ws, size_t ws_size,
                              hipStream_t stream)
{
  const int*           x     = (const int*)d_in[0];
  const unsigned char* mask  = (const unsigned char*)d_in[1];  // jax bool = 1 byte
  const float*         emb   = (const float*)d_in[2];
  const float*         Wh_w  = (const float*)d_in[3];
  const float*         Wh_b  = (const float*)d_in[4];
  const float*         W1_w  = (const float*)d_in[5];
  const float*         W2_w  = (const float*)d_in[6];
  const float*         bvec  = (const float*)d_in[7];
  const float*         cptr  = (const float*)d_in[8];
  const float*         Wf1_w = (const float*)d_in[9];
  const float*         Wf2_w = (const float*)d_in[10];
  const float*         Wf2_b = (const float*)d_in[11];
  const float*         Ws1_w = (const float*)d_in[12];
  const float*         Ws1_b = (const float*)d_in[13];
  const float*         Ws_w  = (const float*)d_in[14];
  const float*         Ws_b  = (const float*)d_in[15];
  float* out = (float*)d_out;

  // ---- workspace carve-up ----
  // fp32 region: 6 x (2048*256) + 3 x (2048*512) floats = 24 MB
  float* ws  = (float*)d_ws;
  float* h   = ws;
  float* h1  = h   + Mrows * Ddim;
  float* h2  = h1  + Mrows * Ddim;
  float* tmp = h2  + Mrows * Ddim;   // h @ Wf2
  float* sfw = tmp + Mrows * Ddim;
  float* sbw = sfw + Mrows * Ddim;
  float* u   = sbw + Mrows * Ddim;   // 2048 x 512 (concat fw|bw)
  float* z   = u   + Mrows * 512;
  float* ats = z   + Mrows * 512;
  // bf16 region (~10 MB)
  __bf16* bf    = (__bf16*)(ats + Mrows * 512);
  __bf16* ebf   = bf;                        // 2048x256
  __bf16* hbf   = ebf   + Mrows * Ddim;
  __bf16* sfwbf = hbf   + Mrows * Ddim;
  __bf16* sbwbf = sfwbf + Mrows * Ddim;
  __bf16* ubf   = sbwbf + Mrows * Ddim;      // 2048x512
  __bf16* zbf   = ubf   + Mrows * 512;
  __bf16* WhT   = zbf   + Mrows * 512;       // 256x256 each
  __bf16* W1T   = WhT   + Ddim * Ddim;
  __bf16* W2T   = W1T   + Ddim * Ddim;
  __bf16* Wf1T  = W2T   + Ddim * Ddim;
  __bf16* Wf2T  = Wf1T  + Ddim * Ddim;
  __bf16* Ws1T  = Wf2T  + Ddim * Ddim;       // 512x512 each
  __bf16* WsT   = Ws1T  + 512 * 512;

  const dim3 blk(256);
  const int gW256 = Ddim * Ddim / 256;   // 256 blocks
  const int gW512 = 512 * 512 / 256;     // 1024 blocks

  // ---- setup: transpose+convert weights, gather+convert embeddings ----
  transpose_bf16<<<gW256, blk, 0, stream>>>(Wh_w,  WhT,  Ddim, Ddim);
  transpose_bf16<<<gW256, blk, 0, stream>>>(W1_w,  W1T,  Ddim, Ddim);
  transpose_bf16<<<gW256, blk, 0, stream>>>(W2_w,  W2T,  Ddim, Ddim);
  transpose_bf16<<<gW256, blk, 0, stream>>>(Wf1_w, Wf1T, Ddim, Ddim);
  transpose_bf16<<<gW256, blk, 0, stream>>>(Wf2_w, Wf2T, Ddim, Ddim);
  transpose_bf16<<<gW512, blk, 0, stream>>>(Ws1_w, Ws1T, 512, 512);
  transpose_bf16<<<gW512, blk, 0, stream>>>(Ws_w,  WsT,  512, 512);
  gather_bf16<<<Mrows, blk, 0, stream>>>(x, emb, ebf);

  const int g256 = (Mrows / 16) * (Ddim / 64) / 8;  // 64 blocks
  const int g512 = (Mrows / 16) * (512 / 64) / 8;   // 128 blocks

  // h = elu(e @ Wh + Wh_b)   (fp32 + bf16 mirror)
  gemm_wmma<EPI_BIAS_ELU, Ddim><<<g256, blk, 0, stream>>>(
      ebf, WhT, Wh_b, nullptr, nullptr, nullptr, h, hbf, Mrows, Ddim, Ddim, 0);
  // h1 = h @ W1 ; h2 = h @ W2 ; tmp = h @ Wf2
  gemm_wmma<EPI_NONE, Ddim><<<g256, blk, 0, stream>>>(
      hbf, W1T, nullptr, nullptr, nullptr, nullptr, h1, nullptr, Mrows, Ddim, Ddim, 0);
  gemm_wmma<EPI_NONE, Ddim><<<g256, blk, 0, stream>>>(
      hbf, W2T, nullptr, nullptr, nullptr, nullptr, h2, nullptr, Mrows, Ddim, Ddim, 0);
  gemm_wmma<EPI_NONE, Ddim><<<g256, blk, 0, stream>>>(
      hbf, Wf2T, nullptr, nullptr, nullptr, nullptr, tmp, nullptr, Mrows, Ddim, Ddim, 0);

  // fused masked-softmax attention: s_fw, s_bw (never materializes att)
  disan_attn<<<Mrows, blk, 0, stream>>>(h, h1, h2, bvec, cptr, mask,
                                        sfw, sbw, sfwbf, sbwbf);

  // u[:, 0:256]   = f*h + (1-f)*s_fw, f = sigmoid(s_fw@Wf1 + tmp + b)
  gemm_wmma<EPI_FUSE_F, Ddim><<<g256, blk, 0, stream>>>(
      sfwbf, Wf1T, Wf2_b, tmp, h, sfw, u, ubf, Mrows, Ddim, 512, 0);
  // u[:, 256:512] = same with s_bw
  gemm_wmma<EPI_FUSE_F, Ddim><<<g256, blk, 0, stream>>>(
      sbwbf, Wf1T, Wf2_b, tmp, h, sbw, u, ubf, Mrows, Ddim, 512, 256);

  // z = elu(u @ Ws1 + Ws1_b) ; att_s = z @ Ws + Ws_b
  gemm_wmma<EPI_BIAS_ELU, 512><<<g512, blk, 0, stream>>>(
      ubf, Ws1T, Ws1_b, nullptr, nullptr, nullptr, z, zbf, Mrows, 512, 512, 0);
  gemm_wmma<EPI_BIAS, 512><<<g512, blk, 0, stream>>>(
      zbf, WsT, Ws_b, nullptr, nullptr, nullptr, ats, nullptr, Mrows, 512, 512, 0);

  // out[b,:] = sum_l u[b,l,:] * att_s[b,l,:]
  disan_reduce<<<Bdim, dim3(512), 0, stream>>>(u, ats, out);
}
